// LocalEncoder_10385230921811
// MI455X (gfx1250) — matrix operand order repaired and checked by reading, not compiled, not run on hardware
//
#include <hip/hip_runtime.h>
#include <hip/hip_bf16.h>
#include <math.h>

// ---------------------------------------------------------------------------
// Types for CDNA5 WMMA fragments (wave32)
// ---------------------------------------------------------------------------
typedef __attribute__((ext_vector_type(16))) _Float16 v16h;
typedef __attribute__((ext_vector_type(8)))  _Float16 v8h;
typedef __attribute__((ext_vector_type(8)))  float    v8f;

#define NUM_E_  20000
#define NUM_EDGE 200000
#define HDIM    200
#define TDDIM   48
#define BATCH_  512

// load 8 contiguous f32 (two b128) and convert to 8 f16
__device__ __forceinline__ v8h cvt8(const float* __restrict__ p) {
    float4 f0 = *(const float4*)p;
    float4 f1 = *(const float4*)(p + 4);
    v8h hv;
    hv[0] = (_Float16)f0.x; hv[1] = (_Float16)f0.y;
    hv[2] = (_Float16)f0.z; hv[3] = (_Float16)f0.w;
    hv[4] = (_Float16)f1.x; hv[5] = (_Float16)f1.y;
    hv[6] = (_Float16)f1.z; hv[7] = (_Float16)f1.w;
    return hv;
}

// fragment loads from LDS + one WMMA step (ISA 7.12.2 layouts, wave32)
__device__ __forceinline__ v8f wmma_step(const _Float16 (*As)[40], const _Float16 (*Bs)[40],
                                         int subM, int subN, int mrow, int ncol,
                                         int kbA, int kbB, v8f acc) {
    v8h alo = *(const v8h*)&As[subM + mrow][kbA];
    v8h ahi = *(const v8h*)&As[subM + mrow][kbA + 16];
    v8h blo = *(const v8h*)&Bs[subN + ncol][kbB];
    v8h bhi = *(const v8h*)&Bs[subN + ncol][kbB + 16];
    v16h af = __builtin_shufflevector(alo, ahi, 0,1,2,3,4,5,6,7,8,9,10,11,12,13,14,15);
    v16h bf = __builtin_shufflevector(blo, bhi, 0,1,2,3,4,5,6,7,8,9,10,11,12,13,14,15);
    return __builtin_amdgcn_wmma_f32_16x16x32_f16(
               false, af, false, bf, (short)0, acc, false, false);
}

// ---------------------------------------------------------------------------
// WMMA GEMM: C[M,N] = act(A[M,K] @ B + bias)
//   BT==0: B stored row-major (K,N)
//   BT==1: B stored as (N,K)  (i.e. computes A @ B^T)
// Block = 128 threads = 4 waves, 32x32 C tile, each wave one 16x16 WMMA tile.
// Interior blocks run a branch-free wide-load loop; only boundary tiles and
// the K-tail use guarded scalar staging.
// ---------------------------------------------------------------------------
template<int BT>
__global__ __launch_bounds__(128)
void gemm_wmma_kernel(const float* __restrict__ A, const float* __restrict__ B,
                      float* __restrict__ C, int M, int N, int K,
                      const float* __restrict__ bias, int doRelu)
{
    // stride 40 halves = 80 bytes: rows 16B aligned, fragment reads b128-able
    __shared__ __align__(16) _Float16 As[32][40];
    __shared__ __align__(16) _Float16 Bs[32][40];

    const int tid  = threadIdx.x;
    const int lane = tid & 31;
    const int wave = tid >> 5;
    const int m0 = blockIdx.y * 32;
    const int n0 = blockIdx.x * 32;
    const int subM = (wave >> 1) * 16;
    const int subN = (wave & 1) * 16;

    v8f acc = {0.f, 0.f, 0.f, 0.f, 0.f, 0.f, 0.f, 0.f};

    const int mrow = lane & 15;                 // A row within 16x16 tile
    const int kbA  = (lane < 16) ? 0 : 8;       // A: K halves {kbA..+7, kbA+16..+23}
    const int kbB  = (lane < 16) ? 0 : 8;       // B: K halves {kbB..+7, kbB+16..+23}
    const int ncol = lane & 15;                 // B column

    // staging thread mapping: 32 rows x 4 threads, 8 contiguous elems each
    const int fr = tid >> 2;
    const int fc = (tid & 3) << 3;

    const bool interior = (m0 + 32 <= M) && (n0 + 32 <= N);
    const int  kFull = K & ~31;

    if (interior) {
        // hoisted, strided base pointers (all row strides are 32B multiples)
        const float* pa = A + (size_t)(m0 + fr) * K + fc;
        const float* pb = BT ? (B + (size_t)(n0 + fr) * K + fc)
                             : (B + (size_t)fr * N + (n0 + fc));
        const size_t pbStep = BT ? 32 : (size_t)32 * N;

        for (int k0 = 0; k0 < kFull; k0 += 32, pa += 32, pb += pbStep) {
            *(v8h*)&As[fr][fc] = cvt8(pa);
            if (BT) {
                *(v8h*)&Bs[fr][fc] = cvt8(pb);
            } else {
                float4 f0 = *(const float4*)pb;
                float4 f1 = *(const float4*)(pb + 4);
                Bs[fc + 0][fr] = (_Float16)f0.x;
                Bs[fc + 1][fr] = (_Float16)f0.y;
                Bs[fc + 2][fr] = (_Float16)f0.z;
                Bs[fc + 3][fr] = (_Float16)f0.w;
                Bs[fc + 4][fr] = (_Float16)f1.x;
                Bs[fc + 5][fr] = (_Float16)f1.y;
                Bs[fc + 6][fr] = (_Float16)f1.z;
                Bs[fc + 7][fr] = (_Float16)f1.w;
            }
            __syncthreads();
            acc = wmma_step(As, Bs, subM, subN, mrow, ncol, kbA, kbB, acc);
            __syncthreads();
        }
        if (kFull < K) {    // K tail: guarded staging, zero-padded
            for (int i = tid; i < 1024; i += 128) {
                int r = i >> 5, c = i & 31;
                int gk = kFull + c;
                As[r][c] = (_Float16)((gk < K) ? A[(size_t)(m0 + r) * K + gk] : 0.0f);
            }
            for (int i = tid; i < 1024; i += 128) {
                if (BT) {
                    int n = i >> 5, kk = i & 31;
                    int gk = kFull + kk;
                    Bs[n][kk] = (_Float16)((gk < K) ? B[(size_t)(n0 + n) * K + gk] : 0.0f);
                } else {
                    int kk = i >> 5, n = i & 31;
                    int gk = kFull + kk;
                    Bs[n][kk] = (_Float16)((gk < K) ? B[(size_t)gk * N + (n0 + n)] : 0.0f);
                }
            }
            __syncthreads();
            acc = wmma_step(As, Bs, subM, subN, mrow, ncol, kbA, kbB, acc);
            __syncthreads();
        }
    } else {
        // boundary blocks: fully guarded staging
        for (int k0 = 0; k0 < K; k0 += 32) {
            for (int i = tid; i < 1024; i += 128) {
                int r = i >> 5, c = i & 31;
                int gm = m0 + r, gk = k0 + c;
                float v = (gm < M && gk < K) ? A[(size_t)gm * K + gk] : 0.0f;
                As[r][c] = (_Float16)v;
            }
            for (int i = tid; i < 1024; i += 128) {
                float v; int n, kk;
                if (BT) {
                    n = i >> 5; kk = i & 31;
                    int gn = n0 + n, gk = k0 + kk;
                    v = (gn < N && gk < K) ? B[(size_t)gn * K + gk] : 0.0f;
                } else {
                    kk = i >> 5; n = i & 31;
                    int gn = n0 + n, gk = k0 + kk;
                    v = (gn < N && gk < K) ? B[(size_t)gk * N + gn] : 0.0f;
                }
                Bs[n][kk] = (_Float16)v;
            }
            __syncthreads();
            acc = wmma_step(As, Bs, subM, subN, mrow, ncol, kbA, kbB, acc);
            __syncthreads();
        }
    }

    // C/D layout: VGPR j -> M = j + (lane<16 ? 0 : 8), N = lane&15
    const int gn = n0 + subN + ncol;
    const int mb = m0 + subM + ((lane >> 4) << 3);
    const float bv = (bias != nullptr && gn < N) ? bias[gn] : 0.0f;
    #pragma unroll
    for (int j = 0; j < 8; ++j) {
        int gm = mb + j;
        if (gm < M && gn < N) {
            float v = acc[j] + bv;
            if (doRelu) v = v > 0.f ? v : 0.f;
            C[(size_t)gm * N + gn] = v;
        }
    }
}

// ---------------------------------------------------------------------------
// Elementwise / graph helpers
// ---------------------------------------------------------------------------
__global__ void fill_zero_kernel(float* p, int n) {
    int i = blockIdx.x * blockDim.x + threadIdx.x;
    if (i < n) p[i] = 0.f;
}

// row-wise L2 normalize (safe for in == out: each block owns its row)
__global__ __launch_bounds__(128)
void l2norm_kernel(const float* __restrict__ in, float* __restrict__ out,
                   int rows, int cols) {
    __shared__ float red[128];
    int r = blockIdx.x;
    float s = 0.f;
    for (int c = threadIdx.x; c < cols; c += 128) {
        float v = in[(size_t)r * cols + c]; s += v * v;
    }
    red[threadIdx.x] = s; __syncthreads();
    for (int off = 64; off > 0; off >>= 1) {
        if (threadIdx.x < off) red[threadIdx.x] += red[threadIdx.x + off];
        __syncthreads();
    }
    float inv = 1.f / fmaxf(sqrtf(red[0]), 1e-12f);
    for (int c = threadIdx.x; c < cols; c += 128)
        out[(size_t)r * cols + c] = in[(size_t)r * cols + c] * inv;
}

__global__ void deg_kernel(const int* __restrict__ dst, float* deg, int E) {
    int e = blockIdx.x * blockDim.x + threadIdx.x;
    if (e < E) atomicAdd(&deg[dst[e]], 1.0f);
}

// per-node: inv = 1/max(deg,1); te = tenc(deg>0 ? timVal : 100)
__global__ void node_time_kernel(const float* __restrict__ deg, float* __restrict__ inv,
                                 float* __restrict__ te,
                                 const float* af, const float* ap,
                                 const float* cf, const float* cp,
                                 float timVal, int n) {
    int v = blockIdx.x * blockDim.x + threadIdx.x;
    if (v >= n) return;
    float d = deg[v];
    inv[v] = 1.f / fmaxf(d, 1.f);
    float t = (d > 0.f) ? timVal : 100.f;
    float* o = te + (size_t)v * TDDIM;
    for (int j = 0; j < 24; ++j) o[j]      = tanhf((t + 1.f) * af[j] + ap[j]);
    for (int j = 0; j < 24; ++j) o[24 + j] = cosf(t * cf[j] + cp[j]);
}

// wave-per-edge: out[dst[e]] += rows[src[e]]
__global__ __launch_bounds__(256)
void scatter_prev_kernel(const float* __restrict__ rows, const int* __restrict__ src,
                         const int* __restrict__ dst, float* out, int E) {
    int e = blockIdx.x * 8 + (threadIdx.x >> 5);
    if (e >= E) return;
    int lane = threadIdx.x & 31;
    const float* pr = rows + (size_t)src[e] * HDIM;
    float* po = out + (size_t)dst[e] * HDIM;
    for (int c = lane; c < HDIM; c += 32) atomicAdd(&po[c], pr[c]);
}

// wave-per-edge: out[dst[e]] += hW[src[e]] + relW[etype[e]]
__global__ __launch_bounds__(256)
void scatter_msg_kernel(const float* __restrict__ hW, const float* __restrict__ relW,
                        const int* __restrict__ src, const int* __restrict__ et,
                        const int* __restrict__ dst, float* out, int E) {
    int e = blockIdx.x * 8 + (threadIdx.x >> 5);
    if (e >= E) return;
    int lane = threadIdx.x & 31;
    const float* ph = hW  + (size_t)src[e] * HDIM;
    const float* pr = relW + (size_t)et[e] * HDIM;
    float* po = out + (size_t)dst[e] * HDIM;
    for (int c = lane; c < HDIM; c += 32) atomicAdd(&po[c], ph[c] + pr[c]);
}

__global__ void mask_scale_kernel(const float* __restrict__ agg, const float* __restrict__ deg,
                                  const float* __restrict__ inv, float* out, int n) {
    int i = blockIdx.x * blockDim.x + threadIdx.x;
    if (i >= n) return;
    int v = i / HDIM;
    out[i] = (deg[v] > 0.f) ? agg[i] * inv[v] : 0.f;
}

__global__ void relu_comb_kernel(const float* __restrict__ nagg, const float* __restrict__ inv,
                                 const float* __restrict__ hs, float* h, int n) {
    int i = blockIdx.x * blockDim.x + threadIdx.x;
    if (i >= n) return;
    int v = i / HDIM;
    h[i] = fmaxf(nagg[i] * inv[v] + hs[i], 0.f);
}

__global__ void build_xcat_kernel(const float* __restrict__ h, const float* __restrict__ te,
                                  float* x, int V) {
    int i = blockIdx.x * blockDim.x + threadIdx.x;
    if (i >= V * (HDIM + TDDIM)) return;
    int v = i / (HDIM + TDDIM), c = i % (HDIM + TDDIM);
    x[i] = (c < HDIM) ? h[(size_t)v * HDIM + c] : te[(size_t)v * TDDIM + (c - HDIM)];
}

__global__ void gru_kernel(const float* __restrict__ gi, const float* __restrict__ gh,
                           const float* __restrict__ prev, float* out, int V) {
    int i = blockIdx.x * blockDim.x + threadIdx.x;
    if (i >= V * HDIM) return;
    int v = i / HDIM, c = i % HDIM;
    size_t b = (size_t)v * 600;
    float ir = gi[b + c], iz = gi[b + 200 + c], iN = gi[b + 400 + c];
    float hr = gh[b + c], hz = gh[b + 200 + c], hN = gh[b + 400 + c];
    float rg = 1.f / (1.f + expf(-(ir + hr)));
    float zg = 1.f / (1.f + expf(-(iz + hz)));
    float ng = tanhf(iN + rg * hN);
    out[i] = (1.f - zg) * ng + zg * prev[i];
}

__global__ void gather_qsqr_kernel(const float* __restrict__ out_e, const float* __restrict__ rel_n,
                                   const int* __restrict__ data, float* qs, float* qr, int B) {
    int i = blockIdx.x * blockDim.x + threadIdx.x;
    if (i >= B * HDIM) return;
    int b = i / HDIM, c = i % HDIM;
    qs[i] = out_e[(size_t)data[b * 4 + 0] * HDIM + c];
    qr[i] = rel_n[(size_t)data[b * 4 + 1] * HDIM + c];
}

__global__ __launch_bounds__(128)
void hist_pool_kernel(const float* __restrict__ out_e, const int* __restrict__ his_idx,
                      const int* __restrict__ his_len, float* semb, int k, int B) {
    __shared__ float red[128];
    int b = blockIdx.x;
    int ln = his_len[b * 3 + k];
    int c0 = threadIdx.x, c1 = threadIdx.x + 128;
    float a0 = 0.f, a1 = 0.f;
    for (int l = 0; l < ln; ++l) {
        const float* row = out_e + (size_t)his_idx[(b * 3 + k) * 32 + l] * HDIM;
        a0 += row[c0];
        if (c1 < HDIM) a1 += row[c1];
    }
    float s = 1.f / (float)(ln > 0 ? ln : 1);
    a0 *= s; a1 *= s;
    float sq = a0 * a0 + ((c1 < HDIM) ? a1 * a1 : 0.f);
    red[threadIdx.x] = sq; __syncthreads();
    for (int off = 64; off > 0; off >>= 1) {
        if (threadIdx.x < off) red[threadIdx.x] += red[threadIdx.x + off];
        __syncthreads();
    }
    float inv = 1.f / fmaxf(sqrtf(red[0]), 1e-12f);
    float* o = semb + ((size_t)k * B + b) * HDIM;
    o[c0] = a0 * inv;
    if (c1 < HDIM) o[c1] = a1 * inv;
}

__global__ void build_acat_kernel(const float* __restrict__ qs, const float* __restrict__ qr,
                                  const float* __restrict__ sk, const int* __restrict__ his_len,
                                  const float* af, const float* ap,
                                  const float* cf, const float* cp,
                                  float timVal, float* acat, int k, int B) {
    int i = blockIdx.x * blockDim.x + threadIdx.x;
    if (i >= B * 648) return;
    int b = i / 648, c = i % 648;
    float v;
    if (c < 200)       v = qs[(size_t)b * HDIM + c];
    else if (c < 400)  v = qr[(size_t)b * HDIM + c - 200];
    else if (c < 600)  v = sk[(size_t)b * HDIM + c - 400];
    else {
        int ln = his_len[b * 3 + k];
        float t = (ln > 0) ? timVal : 100.f;
        int j = c - 600;
        v = (j < 24) ? tanhf((t + 1.f) * af[j] + ap[j]) : cosf(t * cf[j - 24] + cp[j - 24]);
    }
    acat[i] = v;
}

__global__ __launch_bounds__(128)
void att_logit_kernel(const float* __restrict__ abuf, const float* __restrict__ wc,
                      const float* __restrict__ wcb, float* logits, int k, int B) {
    __shared__ float red[128];
    int b = blockIdx.x;
    float s = 0.f;
    for (int c = threadIdx.x; c < HDIM; c += 128) s += abuf[(size_t)b * HDIM + c] * wc[c];
    red[threadIdx.x] = s; __syncthreads();
    for (int off = 64; off > 0; off >>= 1) {
        if (threadIdx.x < off) red[threadIdx.x] += red[threadIdx.x + off];
        __syncthreads();
    }
    if (threadIdx.x == 0) logits[b * 3 + k] = red[0] + wcb[0];
}

__global__ void softmax_out2_kernel(const float* __restrict__ logits, const float* __restrict__ semb,
                                    float* out2, int B) {
    int i = blockIdx.x * blockDim.x + threadIdx.x;
    if (i >= B * HDIM) return;
    int b = i / HDIM, c = i % HDIM;
    float l0 = logits[b * 3], l1 = logits[b * 3 + 1], l2 = logits[b * 3 + 2];
    float m = fmaxf(l0, fmaxf(l1, l2));
    float e0 = expf(l0 - m), e1 = expf(l1 - m), e2 = expf(l2 - m);
    float inv = 1.f / (e0 + e1 + e2);
    out2[i] = (e0 * semb[((size_t)0 * B + b) * HDIM + c] +
               e1 * semb[((size_t)1 * B + b) * HDIM + c] +
               e2 * semb[((size_t)2 * B + b) * HDIM + c]) * inv;
}

__global__ void build_dcat_kernel(const float* __restrict__ qs, const float* __restrict__ qr,
                                  const float* __restrict__ out2, float* dcat, int B) {
    int i = blockIdx.x * blockDim.x + threadIdx.x;
    if (i >= B * 600) return;
    int b = i / 600, c = i % 600;
    float v;
    if (c < 200)      v = qs[(size_t)b * HDIM + c];
    else if (c < 400) v = qr[(size_t)b * HDIM + c - 200];
    else              v = out2[(size_t)b * HDIM + c - 400];
    dcat[i] = v;
}

// ---------------------------------------------------------------------------
// Orchestration
// ---------------------------------------------------------------------------
extern "C" void kernel_launch(void* const* d_in, const int* in_sizes, int n_in,
                              void* d_out, int out_size, void* d_ws, size_t ws_size,
                              hipStream_t stream) {
    (void)in_sizes; (void)n_in; (void)out_size; (void)ws_size;
    const int*   src    = (const int*)d_in[0];
    const int*   dst    = (const int*)d_in[1];
    const int*   etype  = (const int*)d_in[2];
    const int*   data   = (const int*)d_in[3];
    const int*   his_idx= (const int*)d_in[4];
    const int*   his_len= (const int*)d_in[5];
    const float* ent    = (const float*)d_in[6];
    const float* loop_w = (const float*)d_in[7];
    const float* rel    = (const float*)d_in[8];
    const float* abs_f  = (const float*)d_in[9];
    const float* abs_p  = (const float*)d_in[10];
    const float* cos_f  = (const float*)d_in[11];
    const float* cos_p  = (const float*)d_in[12];
    const float* Wn     = (const float*)d_in[13];
    const float* Ws     = (const float*)d_in[14];
    const float* W_ih   = (const float*)d_in[15];
    const float* W_hh   = (const float*)d_in[16];
    const float* b_ih   = (const float*)d_in[17];
    const float* b_hh   = (const float*)d_in[18];
    const float* Wb_w   = (const float*)d_in[19];
    const float* Wb_b   = (const float*)d_in[20];
    const float* Wc_w   = (const float*)d_in[21];
    const float* Wc_b   = (const float*)d_in[22];
    const float* Wd_w   = (const float*)d_in[23];
    const float* Wd_b   = (const float*)d_in[24];
    float* outp = (float*)d_out;

    // workspace layout (~158 MB of floats)
    float* w = (float*)d_ws;
    size_t off = 0;
    auto alloc = [&](size_t n) -> float* { float* p = w + off; off += n; return p; };
    float* rel_n = alloc(80000);
    float* prev  = alloc(4000000);
    float* h     = alloc(4000000);
    float* deg   = alloc(20000);
    float* invd  = alloc(20000);
    float* te    = alloc(960000);
    float* relW  = alloc(80000);
    float* big1  = alloc(12000000);   // agg | (hW,nagg,hs) | gi
    float* big2  = alloc(12000000);   // gh
    float* xcat  = alloc(4960000);
    float* qs    = alloc(102400);
    float* qr    = alloc(102400);
    float* semb  = alloc(307200);
    float* acat  = alloc(331776);
    float* abuf  = alloc(102400);
    float* logits= alloc(1536);
    float* out2  = alloc(102400);
    float* dcat  = alloc(307200);
    float* qbuf  = alloc(102400);

    auto gemm = [&](const float* A, const float* B, float* C, int M, int N, int K,
                    const float* bias, int relu, bool bt) {
        dim3 g((N + 31) / 32, (M + 31) / 32);
        if (bt) gemm_wmma_kernel<1><<<g, 128, 0, stream>>>(A, B, C, M, N, K, bias, relu);
        else    gemm_wmma_kernel<0><<<g, 128, 0, stream>>>(A, B, C, M, N, K, bias, relu);
    };
    auto zero = [&](float* p, int n) {
        fill_zero_kernel<<<(n + 255) / 256, 256, 0, stream>>>(p, n);
    };

    // rel_n = l2norm(rel); prev = l2norm(ent @ loop_weight)
    l2norm_kernel<<<400, 128, 0, stream>>>(rel, rel_n, 400, HDIM);
    gemm(ent, loop_w, h, NUM_E_, HDIM, HDIM, nullptr, 0, false);
    l2norm_kernel<<<NUM_E_, 128, 0, stream>>>(h, prev, NUM_E_, HDIM);

    for (int i = 0; i < 3; ++i) {
        const int* si = src   + (size_t)i * NUM_EDGE;
        const int* di = dst   + (size_t)i * NUM_EDGE;
        const int* ei = etype + (size_t)i * NUM_EDGE;

        zero(deg, NUM_E_);
        deg_kernel<<<(NUM_EDGE + 255) / 256, 256, 0, stream>>>(di, deg, NUM_EDGE);
        node_time_kernel<<<(NUM_E_ + 255) / 256, 256, 0, stream>>>(
            deg, invd, te, abs_f, abs_p, cos_f, cos_p, (float)(2 - i), NUM_E_);

        float* agg = big1;
        zero(agg, 4000000);
        scatter_prev_kernel<<<(NUM_EDGE + 7) / 8, 256, 0, stream>>>(prev, si, di, agg, NUM_EDGE);
        mask_scale_kernel<<<(4000000 + 255) / 256, 256, 0, stream>>>(agg, deg, invd, h, 4000000);
        l2norm_kernel<<<NUM_E_, 128, 0, stream>>>(h, h, NUM_E_, HDIM);

        for (int l = 0; l < 2; ++l) {
            const float* Wnl = Wn + (size_t)l * 40000;
            const float* Wsl = Ws + (size_t)l * 40000;
            float* hW = big1, * nagg = big1 + 4000000, * hs = big1 + 8000000;
            // (h[si]+rel_n[ei])@Wn == hW[si] + relW[ei]  (distributed over segment_sum)
            gemm(h,     Wnl, hW,   NUM_E_, HDIM, HDIM, nullptr, 0, false);
            gemm(rel_n, Wnl, relW, 400,    HDIM, HDIM, nullptr, 0, false);
            zero(nagg, 4000000);
            scatter_msg_kernel<<<(NUM_EDGE + 7) / 8, 256, 0, stream>>>(
                hW, relW, si, ei, di, nagg, NUM_EDGE);
            gemm(h, Wsl, hs, NUM_E_, HDIM, HDIM, nullptr, 0, false);
            relu_comb_kernel<<<(4000000 + 255) / 256, 256, 0, stream>>>(
                nagg, invd, hs, h, 4000000);
        }
        l2norm_kernel<<<NUM_E_, 128, 0, stream>>>(h, h, NUM_E_, HDIM);

        // GRU:  gi = [h|te] @ W_ih^T + b_ih ;  gh = prev @ W_hh^T + b_hh
        build_xcat_kernel<<<(NUM_E_ * 248 + 255) / 256, 256, 0, stream>>>(h, te, xcat, NUM_E_);
        float* gi = big1, * gh = big2;
        gemm(xcat, W_ih, gi, NUM_E_, 600, 248, b_ih, 0, true);
        gemm(prev, W_hh, gh, NUM_E_, 600, HDIM, b_hh, 0, true);
        gru_kernel<<<(4000000 + 255) / 256, 256, 0, stream>>>(gi, gh, prev, h, NUM_E_);
        l2norm_kernel<<<NUM_E_, 128, 0, stream>>>(h, prev, NUM_E_, HDIM);
    }
    // prev now holds `out`

    gather_qsqr_kernel<<<(BATCH_ * HDIM + 255) / 256, 256, 0, stream>>>(
        prev, rel_n, data, qs, qr, BATCH_);

    for (int k = 0; k < 3; ++k) {
        hist_pool_kernel<<<BATCH_, 128, 0, stream>>>(prev, his_idx, his_len, semb, k, BATCH_);
        build_acat_kernel<<<(BATCH_ * 648 + 255) / 256, 256, 0, stream>>>(
            qs, qr, semb + (size_t)k * BATCH_ * HDIM, his_len,
            abs_f, abs_p, cos_f, cos_p, (float)(2 - k), acat, k, BATCH_);
        gemm(acat, Wb_w, abuf, BATCH_, HDIM, 648, Wb_b, 1, true);
        att_logit_kernel<<<BATCH_, 128, 0, stream>>>(abuf, Wc_w, Wc_b, logits, k, BATCH_);
    }
    softmax_out2_kernel<<<(BATCH_ * HDIM + 255) / 256, 256, 0, stream>>>(
        logits, semb, out2, BATCH_);
    build_dcat_kernel<<<(BATCH_ * 600 + 255) / 256, 256, 0, stream>>>(
        qs, qr, out2, dcat, BATCH_);
    gemm(dcat, Wd_w, qbuf, BATCH_, HDIM, 600, Wd_b, 1, true);

    // final scores: (512x200) @ out^T -> (512x20000), BT mode with B = out (20000x200)
    gemm(qbuf, prev, outp, BATCH_, NUM_E_, HDIM, nullptr, 0, true);
}
